// SupernodePooling_66219805770008
// MI455X (gfx1250) — matrix operand order repaired and checked by reading, not compiled, not run on hardware
//
#include <hip/hip_runtime.h>
#include <hip/hip_bf16.h>

// ---------------------------------------------------------------------------
// SupernodePooling for MI455X (gfx1250, wave32, WMMA)
//
//  - Kernel 0: convert + SWIZZLE W1 [384x192], W2 [192x192] fp32 -> fp16 into
//    d_ws, laid out exactly as the WMMA B-operand register layout so the main
//    kernel fetches each 32x16 B tile as one contiguous 32B load per lane.
//  - Kernel 1: one workgroup (256 thr = 8 wave32) per supernode:
//      * stage 33 node records (feat3+pos3), W_in/b_in, freq table in LDS
//      * compute msg embeddings directly into LDS in WMMA A-operand swizzled
//        layout (fp16)
//      * layer1: 32x384 @ 384x192 via v_wmma_f32_16x16x32_f16 (+b1, branchless
//        exact-GELU via A&S erf), result re-swizzled to LDS as layer-2 A
//      * layer2: 32x192 @ 192x192 via WMMA
//      * column-mean over the 32 edges via per-lane sums + ds_add_f32
//
// WMMA register layouts implemented (per CDNA5 ISA 7.12.2):
//  A 16x32 f16: lane<16 holds row=lane, K={0..7,16..23}; lane>=16 holds
//               row=lane-16, K={8..15,24..31}; halves packed [lo8|hi8].
//  B 32x16 f16: lane<16 holds col=lane, K={0..7,16..23}; lane>=16 col=lane-16,
//               K={8..15,24..31}; halves packed [lo8|hi8] (same as A).
//  C 16x16 f32: VGPR v, lane<16 -> (M=v, N=lane); lane>=16 -> (M=8+v, N=lane-16).
// ---------------------------------------------------------------------------

#define N_HID   192
#define N_2H    384
#define DEG     32
#define BLK     256
#define KT1     12          // 384/32 K-tiles, layer 1
#define KT2     6           // 192/32 K-tiles, layer 2
#define NT      12          // 192/16 N-tiles

typedef __attribute__((ext_vector_type(16))) _Float16 v16h;
typedef __attribute__((ext_vector_type(8)))  float    v8f;

// ---- swizzle helpers (element -> flat half index) --------------------------
// A-operand tile storage: [mtile][ktile][lane32][half16]
__device__ __forceinline__ int a_swz(int row /*0..31*/, int col /*0..inf*/,
                                     int ktiles) {
    const int m   = row >> 4;
    const int r15 = row & 15;
    const int kt  = col >> 5;
    const int kk  = col & 31;
    const int lane = (((kk >> 3) & 1) << 4) | r15;
    const int h    = (kk & 7) | ((kk >> 4) << 3);
    return (((m * ktiles + kt) * 32) + lane) * 16 + h;
}
// B-operand tile storage: [ktile][ntile][lane32][half16]
__device__ __forceinline__ int b_swz(int K, int col) {
    const int kt  = K >> 5;
    const int kkb = K & 31;
    const int nt  = col >> 4;
    const int c15 = col & 15;
    const int lane = (((kkb >> 3) & 1) << 4) | c15;
    const int h    = (kkb & 7) | ((kkb >> 4) << 3);
    return (((kt * NT + nt) * 32) + lane) * 16 + h;
}

// branchless exact-GELU: erf via Abramowitz-Stegun 7.1.26 (|err| < 1.5e-7)
__device__ __forceinline__ float gelu_exact(float x) {
    const float z  = x * 0.70710678118654752f;
    const float az = fabsf(z);
    const float t  = 1.0f / (1.0f + 0.3275911f * az);
    const float poly = t * (0.254829592f + t * (-0.284496736f +
                       t * (1.421413741f + t * (-1.453152027f +
                       t * 1.061405429f))));
    const float erfv = copysignf(1.0f - poly * __expf(-az * az), z);
    return 0.5f * x * (1.0f + erfv);
}

__global__ __launch_bounds__(BLK)
void snp_convert_weights(const float* __restrict__ W1f,
                         const float* __restrict__ W2f,
                         _Float16* __restrict__ gW1s,
                         _Float16* __restrict__ gW2s) {
    int i = blockIdx.x * blockDim.x + threadIdx.x;
    if (i < N_2H * N_HID) {
        const int K = i / N_HID, col = i - K * N_HID;
        gW1s[b_swz(K, col)] = (_Float16)W1f[i];
    }
    if (i < N_HID * N_HID) {
        const int K = i / N_HID, col = i - K * N_HID;
        gW2s[b_swz(K, col)] = (_Float16)W2f[i];
    }
}

__global__ __launch_bounds__(BLK)
void snp_main(const float* __restrict__ feat,      // [N,3]
              const float* __restrict__ pos,       // [N,3]
              const int*   __restrict__ src_idx,   // [E]
              const int*   __restrict__ dst_idx,   // [E]
              const float* __restrict__ W_in,      // [3,192]
              const float* __restrict__ b_in,      // [192]
              const _Float16* __restrict__ gW1s,   // swizzled [12][12][32][16]
              const float* __restrict__ b1,        // [192]
              const _Float16* __restrict__ gW2s,   // swizzled [6][12][32][16]
              const float* __restrict__ b2,        // [192]
              float* __restrict__ out)             // [S,192]
{
    __shared__ alignas(32) _Float16 sA [2 * KT1 * 32 * 16]; // layer1 A, swizzled
    __shared__ alignas(32) _Float16 sH2[2 * KT2 * 32 * 16]; // layer2 A, swizzled
    __shared__ float sSum[N_HID];
    __shared__ float sFreq[32];
    __shared__ float sNode[33][6];          // feat[0..2], pos[0..2] per row
    __shared__ float sWin[3 * N_HID];       // W_in staged
    __shared__ float sBin[N_HID];           // b_in staged

    const int s    = blockIdx.x;
    const int tid  = threadIdx.x;
    const int lane = tid & 31;
    const int wave = tid >> 5;
    const int half = lane >> 4;
    const int l15  = lane & 15;

    // ---- Phase 0: stage constants + node records --------------------------
    for (int c = tid; c < N_HID; c += BLK) { sSum[c] = 0.0f; sBin[c] = b_in[c]; }
    for (int c = tid; c < 3 * N_HID; c += BLK) sWin[c] = W_in[c];
    if (tid < 32)
        sFreq[tid] = __powf(10000.0f, -(float)(2 * tid) * (1.0f / 64.0f));
    const int dstNode = dst_idx[s * DEG];
    for (int idx = tid; idx < 33 * 6; idx += BLK) {
        const int r = idx / 6, c = idx - r * 6;
        const int node = (r < DEG) ? src_idx[s * DEG + r] : dstNode;
        sNode[r][c] = (c < 3) ? feat[node * 3 + c] : pos[node * 3 + (c - 3)];
    }
    __syncthreads();

    // ---- Phase 1: embeddings -> swizzled fp16 A operand -------------------
    // logical rr 0..31: src emb -> cols 0..191 ; rr 32..63: dst emb -> 192..383
    for (int idx = tid; idx < 64 * N_HID; idx += BLK) {
        const int rr  = idx / N_HID;
        const int d   = idx - rr * N_HID;
        const int row = rr & 31;
        const int nr  = (rr < DEG) ? rr : 32;           // node record
        const int col = d + ((rr < DEG) ? 0 : N_HID);
        const int axis = d / 64;
        const int j    = d - axis * 64;                 // [sin 0..31 | cos 32..63]
        const int k    = (j < 32) ? j : (j - 32);
        const float ang  = sNode[nr][3 + axis] * sFreq[k];
        const float trig = (j < 32) ? __sinf(ang) : __cosf(ang);
        const float v = sBin[d] + trig
                      + sNode[nr][0] * sWin[d]
                      + sNode[nr][1] * sWin[N_HID + d]
                      + sNode[nr][2] * sWin[2 * N_HID + d];
        sA[a_swz(row, col, KT1)] = (_Float16)v;
    }
    __syncthreads();

    // ---- wave tiling: m = M-tile (16 rows), 3 N-tiles per wave ------------
    const int m  = wave & 1;
    const int n0 = (wave >> 1) * 3;     // first of 3 N-tiles

    // ---- Phase 2: layer1 + bias + GELU -> swizzled layer-2 A --------------
    {
        v8f acc[3] = {{}, {}, {}};
        for (int kt = 0; kt < KT1; ++kt) {
            const v16h a = *(const v16h*)&sA[((m * KT1 + kt) * 32 + lane) * 16];
#pragma unroll
            for (int nt = 0; nt < 3; ++nt) {
                const v16h b = *(const v16h*)
                    &gW1s[((kt * NT + n0 + nt) * 32 + lane) * 16];
                acc[nt] = __builtin_amdgcn_wmma_f32_16x16x32_f16(
                              false, a, false, b, (short)0, acc[nt], false, false);
            }
        }
#pragma unroll
        for (int nt = 0; nt < 3; ++nt) {
            const int col = (n0 + nt) * 16 + l15;
            const float bb = b1[col];
#pragma unroll
            for (int v = 0; v < 8; ++v) {
                const int row = (m << 4) + (half << 3) + v;
                sH2[a_swz(row, col, KT2)] = (_Float16)gelu_exact(acc[nt][v] + bb);
            }
        }
    }
    __syncthreads();

    // ---- Phase 3: layer2 + pooled column sums -----------------------------
    {
        v8f acc[3] = {{}, {}, {}};
        for (int kt = 0; kt < KT2; ++kt) {
            const v16h a = *(const v16h*)&sH2[((m * KT2 + kt) * 32 + lane) * 16];
#pragma unroll
            for (int nt = 0; nt < 3; ++nt) {
                const v16h b = *(const v16h*)
                    &gW2s[((kt * NT + n0 + nt) * 32 + lane) * 16];
                acc[nt] = __builtin_amdgcn_wmma_f32_16x16x32_f16(
                              false, a, false, b, (short)0, acc[nt], false, false);
            }
        }
#pragma unroll
        for (int nt = 0; nt < 3; ++nt) {
            float colsum = 0.0f;
#pragma unroll
            for (int v = 0; v < 8; ++v) colsum += acc[nt][v];
            atomicAdd(&sSum[(n0 + nt) * 16 + l15], colsum);   // ds_add_f32
        }
    }
    __syncthreads();

    // ---- Phase 4: mean over 32 edges + b2 ---------------------------------
    for (int c = tid; c < N_HID; c += BLK)
        out[s * N_HID + c] = sSum[c] * (1.0f / (float)DEG) + b2[c];
}

extern "C" void kernel_launch(void* const* d_in, const int* in_sizes, int n_in,
                              void* d_out, int out_size, void* d_ws, size_t ws_size,
                              hipStream_t stream) {
    const float* feat    = (const float*)d_in[0];
    const float* pos     = (const float*)d_in[1];
    const int*   src_idx = (const int*)d_in[2];
    const int*   dst_idx = (const int*)d_in[3];
    const float* W_in    = (const float*)d_in[4];
    const float* b_in    = (const float*)d_in[5];
    const float* W1f     = (const float*)d_in[6];
    const float* b1      = (const float*)d_in[7];
    const float* W2f     = (const float*)d_in[8];
    const float* b2      = (const float*)d_in[9];
    float* out = (float*)d_out;

    const int E = in_sizes[3];
    const int S = E / DEG;          // 8192 supernodes

    // swizzled fp16 weights in workspace: W1 then W2 -> 221184 B total
    _Float16* gW1s = (_Float16*)d_ws;
    _Float16* gW2s = gW1s + N_2H * N_HID;

    {
        const int n = N_2H * N_HID;
        snp_convert_weights<<<(n + BLK - 1) / BLK, BLK, 0, stream>>>(
            W1f, W2f, gW1s, gW2s);
    }

    snp_main<<<S, BLK, 0, stream>>>(feat, pos, src_idx, dst_idx,
                                    W_in, b_in, gW1s, b1, gW2s, b2, out);
}
